// MoE_62603443306661
// MI455X (gfx1250) — matrix operand order repaired and checked by reading, not compiled, not run on hardware
//
#include <hip/hip_runtime.h>

#define DEV_INLINE __device__ __forceinline__

typedef __attribute__((ext_vector_type(16))) __bf16 v16bf;
typedef __attribute__((ext_vector_type(8)))  __bf16 bf16x8;
typedef __attribute__((ext_vector_type(8)))  float  v8f;

namespace {
constexpr int DIMC   = 1024;
constexpr int HIDC   = 2732;
constexpr int EC     = 8;
constexpr int NTOK   = 2 * 2048;          // 4096 tokens
constexpr int HP     = 2816;              // HID padded to multiple of 256
constexpr int CAP    = NTOK;              // worst-case tokens per expert
constexpr int MT     = 32;                // tokens per block (2 row-tiles)
constexpr int HCHUNK = 256;               // H per fused iteration (16 waves x 16)
constexpr int NIT    = HP / HCHUNK;       // 11
}

union FragB { v16bf v; bf16x8 h[2]; };

DEV_INLINE unsigned short f2bf(float f) {
  unsigned int u = __float_as_uint(f);
  u += 0x7FFFu + ((u >> 16) & 1u);        // round-to-nearest-even
  return (unsigned short)(u >> 16);
}
DEV_INLINE unsigned int pack2(float a, float b) {
  return (unsigned int)f2bf(a) | ((unsigned int)f2bf(b) << 16);
}

// ---------------------------------------------------------------- zero
__global__ void zero_kernel(float4* __restrict__ out4, int n4, int* __restrict__ cnt) {
  int i = blockIdx.x * blockDim.x + threadIdx.x;
  float4 z = make_float4(0.f, 0.f, 0.f, 0.f);
  for (int q = i; q < n4; q += gridDim.x * blockDim.x) out4[q] = z;
  if (i < EC) cnt[i] = 0;
}

// --------------------------------------------- fp32 -> bf16 conversions
// src: [E, HID, DIM] -> dst: [E, HP, DIM], zero-padded rows HID..HP
__global__ void cvt_rows_kernel(const float* __restrict__ src, unsigned short* __restrict__ dst) {
  const int total = EC * HP * (DIMC / 4);
  for (int q = blockIdx.x * blockDim.x + threadIdx.x; q < total; q += gridDim.x * blockDim.x) {
    int row = q >> 8;                 // DIM/4 == 256
    int c4  = (q & 255) << 2;
    int e   = row / HP;
    int h   = row - e * HP;
    float4 v = make_float4(0.f, 0.f, 0.f, 0.f);
    if (h < HIDC) v = *(const float4*)(src + ((size_t)e * HIDC + h) * DIMC + c4);
    *(uint2*)(dst + (size_t)row * DIMC + c4) = make_uint2(pack2(v.x, v.y), pack2(v.z, v.w));
  }
}

// src: [E*DIM, HID] -> dst: [E*DIM, HP], zero-padded cols HID..HP
__global__ void cvt_cols_kernel(const float* __restrict__ src, unsigned short* __restrict__ dst) {
  const int total = EC * DIMC * (HP / 4);
  for (int q = blockIdx.x * blockDim.x + threadIdx.x; q < total; q += gridDim.x * blockDim.x) {
    int row = q / (HP / 4);
    int c4  = (q - row * (HP / 4)) << 2;
    float4 v = make_float4(0.f, 0.f, 0.f, 0.f);
    if (c4 < HIDC) v = *(const float4*)(src + (size_t)row * HIDC + c4);   // HID % 4 == 0
    *(uint2*)(dst + (size_t)row * HP + c4) = make_uint2(pack2(v.x, v.y), pack2(v.z, v.w));
  }
}

// x: [NTOK, DIM] fp32 -> bf16
__global__ void cvt_x_kernel(const float* __restrict__ src, unsigned short* __restrict__ dst) {
  const int total = NTOK * (DIMC / 4);
  for (int q = blockIdx.x * blockDim.x + threadIdx.x; q < total; q += gridDim.x * blockDim.x) {
    const float4 v = *(const float4*)(src + (size_t)q * 4);
    *(uint2*)(dst + (size_t)q * 4) = make_uint2(pack2(v.x, v.y), pack2(v.z, v.w));
  }
}

// ---------------------------------------------------------------- gate
__global__ __launch_bounds__(256)
void gate_kernel(const float* __restrict__ x, const float* __restrict__ Wg,
                 int* __restrict__ cnt, int* __restrict__ tokidx, float* __restrict__ tokw) {
  const int wave = threadIdx.x >> 5;
  const int lane = threadIdx.x & 31;
  const int n = blockIdx.x * 8 + wave;
  if (n >= NTOK) return;

  float acc[EC];
#pragma unroll
  for (int e = 0; e < EC; ++e) acc[e] = 0.f;

  const float* xr = x + (size_t)n * DIMC;
  for (int k = lane; k < DIMC; k += 32) {
    float xv = xr[k];
#pragma unroll
    for (int e = 0; e < EC; ++e) acc[e] += xv * Wg[e * DIMC + k];
  }
#pragma unroll
  for (int e = 0; e < EC; ++e) {
    acc[e] += __shfl_xor(acc[e], 16, 32);
    acc[e] += __shfl_xor(acc[e],  8, 32);
    acc[e] += __shfl_xor(acc[e],  4, 32);
    acc[e] += __shfl_xor(acc[e],  2, 32);
    acc[e] += __shfl_xor(acc[e],  1, 32);
  }
  if (lane == 0) {
    int e0 = 0; float s0 = acc[0];
#pragma unroll
    for (int e = 1; e < EC; ++e) if (acc[e] > s0) { s0 = acc[e]; e0 = e; }
    int e1 = (e0 == 0) ? 1 : 0; float s1 = -3.4e38f;
#pragma unroll
    for (int e = 0; e < EC; ++e) if (e != e0 && acc[e] > s1) { s1 = acc[e]; e1 = e; }
    float w0 = 1.0f / (1.0f + __expf(s1 - s0));   // softmax over top-2 (s0 >= s1)
    float w1 = 1.0f - w0;
    int slot0 = atomicAdd(&cnt[e0], 1);
    tokidx[e0 * CAP + slot0] = n;  tokw[e0 * CAP + slot0] = w0;
    int slot1 = atomicAdd(&cnt[e1], 1);
    tokidx[e1 * CAP + slot1] = n;  tokw[e1 * CAP + slot1] = w1;
  }
}

// ------------------------------------------------ fused expert SwiGLU
// grid: (CAP/MT token-tiles, E experts); block: 512 threads = 16 waves.
// Each wave: stage1 computes a 32x16 h tile; stage2 owns 64 output columns.
__global__ __launch_bounds__(512)
void moe_expert_kernel(const __bf16* __restrict__ xb,
                       const __bf16* __restrict__ W1b,
                       const __bf16* __restrict__ W3b,
                       const __bf16* __restrict__ W2b,
                       const int* __restrict__ cnt,
                       const int* __restrict__ tokidx,
                       const float* __restrict__ tokw,
                       float* __restrict__ out) {
  const int e = blockIdx.y;
  const int nassigned = cnt[e];
  const int tile0 = blockIdx.x * MT;
  if (tile0 >= nassigned) return;                 // uniform per block

  __shared__ __align__(16) unsigned short sh[MT][HCHUNK + 8];  // ~16.9 KB h chunk (bf16)
  __shared__ int   stok[MT];
  __shared__ float sgw[MT];

  const int tid  = threadIdx.x;
  const int wave = tid >> 5;
  const int lane = tid & 31;

  if (tid < MT) {
    int slot = tile0 + tid;
    if (slot < nassigned) {
      stok[tid] = tokidx[e * CAP + slot];
      sgw[tid]  = tokw[e * CAP + slot];
    } else {
      stok[tid] = tokidx[e * CAP];    // clamp to a valid token, weight 0
      sgw[tid]  = 0.0f;
    }
  }
  __syncthreads();

  const int col   = lane & 15;
  const int abase = (lane & 16);         // A frag: K sub-offset {0,16}
  const int bbase = (lane & 16);         // B frag: K sub-offset {0,16}

  // A-fragment row pointers: gathered bf16 x rows (WGP$-resident working set)
  const __bf16* xr0 = xb + (size_t)stok[col]      * DIMC;
  const __bf16* xr1 = xb + (size_t)stok[16 + col] * DIMC;
  const unsigned short* shr0 = &sh[col][0];
  const unsigned short* shr1 = &sh[16 + col][0];

  const int dcol0 = wave * 64;           // this wave's 64 output columns of D
  const __bf16* w2r[4];
#pragma unroll
  for (int t = 0; t < 4; ++t)
    w2r[t] = W2b + ((size_t)e * DIMC + dcol0 + t * 16 + col) * HP;

  const v8f vz = {0.f, 0.f, 0.f, 0.f, 0.f, 0.f, 0.f, 0.f};
  v8f accy[8];                           // [rowtile*4 + coltile]
#pragma unroll
  for (int t = 0; t < 8; ++t) accy[t] = vz;

  for (int it = 0; it < NIT; ++it) {
    // ---- stage 1: 32x16 h tile over full K = 1024; B reused for 2 row-tiles ----
    const int hrow = it * HCHUNK + wave * 16 + col;
    const __bf16* w1row = W1b + ((size_t)e * HP + hrow) * DIMC;
    const __bf16* w3row = W3b + ((size_t)e * HP + hrow) * DIMC;

    v8f a10 = vz, a11 = vz, a30 = vz, a31 = vz;
#pragma unroll 4
    for (int kk = 0; kk < DIMC / 32; ++kk) {
      FragB a0, a1, b1, b3;
      const int ka = kk * 32 + abase;
      const int kb = kk * 32 + bbase;
      a0.h[0] = *(const bf16x8*)(xr0 + ka);
      a0.h[1] = *(const bf16x8*)(xr0 + ka + 8);
      a1.h[0] = *(const bf16x8*)(xr1 + ka);
      a1.h[1] = *(const bf16x8*)(xr1 + ka + 8);
      b1.h[0] = *(const bf16x8*)(w1row + kb);
      b1.h[1] = *(const bf16x8*)(w1row + kb + 8);
      b3.h[0] = *(const bf16x8*)(w3row + kb);
      b3.h[1] = *(const bf16x8*)(w3row + kb + 8);
      a10 = __builtin_amdgcn_wmma_f32_16x16x32_bf16(false, a0.v, false, b1.v, (short)0, a10, false, false);
      a11 = __builtin_amdgcn_wmma_f32_16x16x32_bf16(false, a1.v, false, b1.v, (short)0, a11, false, false);
      a30 = __builtin_amdgcn_wmma_f32_16x16x32_bf16(false, a0.v, false, b3.v, (short)0, a30, false, false);
      a31 = __builtin_amdgcn_wmma_f32_16x16x32_bf16(false, a1.v, false, b3.v, (short)0, a31, false, false);
    }

    // silu(h1) * h3 -> bf16 -> LDS (fast v_rcp instead of IEEE divide)
    const int rb = (lane >> 4) << 3;
    const int hc = wave * 16 + col;
#pragma unroll
    for (int i = 0; i < 8; ++i) {
      float u0 = a10[i];
      float u1 = a11[i];
      float h0 = u0 * __builtin_amdgcn_rcpf(1.0f + __expf(-u0)) * a30[i];
      float h1 = u1 * __builtin_amdgcn_rcpf(1.0f + __expf(-u1)) * a31[i];
      sh[rb + i][hc]      = f2bf(h0);
      sh[16 + rb + i][hc] = f2bf(h1);
    }
    __syncthreads();

    // ---- stage 2: accy += h_chunk[32x256] @ W2_chunk[64x256]^T ----
#pragma unroll 2
    for (int kk = 0; kk < HCHUNK / 32; ++kk) {
      FragB a0, a1;
      const int ka = kk * 32 + abase;
      a0.h[0] = *(const bf16x8*)(shr0 + ka);
      a0.h[1] = *(const bf16x8*)(shr0 + ka + 8);
      a1.h[0] = *(const bf16x8*)(shr1 + ka);
      a1.h[1] = *(const bf16x8*)(shr1 + ka + 8);
      const int kb = it * HCHUNK + kk * 32 + bbase;
#pragma unroll
      for (int t = 0; t < 4; ++t) {
        FragB b;
        b.h[0] = *(const bf16x8*)(w2r[t] + kb);
        b.h[1] = *(const bf16x8*)(w2r[t] + kb + 8);
        accy[t]     = __builtin_amdgcn_wmma_f32_16x16x32_bf16(false, a0.v, false, b.v, (short)0, accy[t],     false, false);
        accy[4 + t] = __builtin_amdgcn_wmma_f32_16x16x32_bf16(false, a1.v, false, b.v, (short)0, accy[4 + t], false, false);
      }
    }
    __syncthreads();
  }

  // epilogue: weighted scatter-add (each token contributes to exactly 2 experts)
  const int rb = (lane >> 4) << 3;
#pragma unroll
  for (int r = 0; r < 2; ++r) {
#pragma unroll
    for (int t = 0; t < 4; ++t) {
      const int d = dcol0 + t * 16 + col;
#pragma unroll
      for (int i = 0; i < 8; ++i) {
        const int m = r * 16 + rb + i;
        atomicAdd(out + (size_t)stok[m] * DIMC + d, sgw[m] * accy[r * 4 + t][i]);
      }
    }
  }
}

// ---------------------------------------------------------------- launch
extern "C" void kernel_launch(void* const* d_in, const int* in_sizes, int n_in,
                              void* d_out, int out_size, void* d_ws, size_t ws_size,
                              hipStream_t stream) {
  (void)in_sizes; (void)n_in; (void)out_size; (void)ws_size;
  const float* x  = (const float*)d_in[0];
  const float* Wg = (const float*)d_in[1];
  const float* W1 = (const float*)d_in[2];
  const float* W2 = (const float*)d_in[3];
  const float* W3 = (const float*)d_in[4];
  float* out = (float*)d_out;

  char* ws = (char*)d_ws;
  const size_t szW = (size_t)EC * HP * DIMC * sizeof(unsigned short);  // ~46 MB each
  unsigned short* W1b = (unsigned short*)(ws);
  unsigned short* W3b = (unsigned short*)(ws + szW);
  unsigned short* W2b = (unsigned short*)(ws + 2 * szW);
  int*   cnt    = (int*)  (ws + 3 * szW);
  int*   tokidx = (int*)  (ws + 3 * szW + 256);
  float* tokw   = (float*)(ws + 3 * szW + 256 + (size_t)EC * CAP * sizeof(int));
  unsigned short* xbf = (unsigned short*)(ws + 3 * szW + 256 +
                                          2 * (size_t)EC * CAP * sizeof(int));

  zero_kernel<<<2048, 256, 0, stream>>>((float4*)out, NTOK * DIMC / 4, cnt);
  cvt_rows_kernel<<<8192, 256, 0, stream>>>(W1, W1b);
  cvt_rows_kernel<<<8192, 256, 0, stream>>>(W3, W3b);
  cvt_cols_kernel<<<8192, 256, 0, stream>>>(W2, W2b);
  cvt_x_kernel<<<2048, 256, 0, stream>>>(x, xbf);
  gate_kernel<<<NTOK / 8, 256, 0, stream>>>(x, Wg, cnt, tokidx, tokw);

  dim3 grid(CAP / MT, EC);
  moe_expert_kernel<<<grid, 512, 0, stream>>>((const __bf16*)xbf,
      (const __bf16*)W1b, (const __bf16*)W3b, (const __bf16*)W2b,
      cnt, tokidx, tokw, out);
}
